// GRU_20959440405111
// MI455X (gfx1250) — matrix operand order repaired and checked by reading, not compiled, not run on hardware
//
#include <hip/hip_runtime.h>
#include <hip/hip_bf16.h>

// ---------------------------------------------------------------------------
// Problem constants (from reference): B=256, T=128, D=512, H=1024
// ---------------------------------------------------------------------------
#define BB   256
#define TT   128
#define DD   512
#define HH   1024
#define H2   2048
#define H3   3072
#define BT   (BB * TT)           // 32768

typedef __attribute__((ext_vector_type(16))) __bf16 v16bf;
typedef __attribute__((ext_vector_type(8)))  float  v8f;

union ABFrag { v16bf v; uint4 q[2]; };

// ---------------------------------------------------------------------------
// fp32 -> bf16 round-to-nearest-even
// ---------------------------------------------------------------------------
__device__ __forceinline__ unsigned short f32_to_bf16(float f) {
    unsigned u = __float_as_uint(f);
    unsigned r = u + 0x7fffu + ((u >> 16) & 1u);
    if ((u & 0x7f800000u) == 0x7f800000u) r = u;   // Inf/NaN passthrough
    return (unsigned short)(r >> 16);
}

__device__ __forceinline__ float sigmoidf_(float x) {
    return 1.0f / (1.0f + __expf(-x));
}

// block-wide sum over 256 threads (8 waves of 32)
__device__ __forceinline__ float block_sum(float v, float* sh8) {
    #pragma unroll
    for (int o = 16; o > 0; o >>= 1) v += __shfl_xor(v, o, 32);
    int lane = threadIdx.x & 31;
    int w    = threadIdx.x >> 5;
    if (lane == 0) sh8[w] = v;
    __syncthreads();
    float r = sh8[0];
    #pragma unroll
    for (int i = 1; i < 8; ++i) r += sh8[i];
    __syncthreads();
    return r;
}

// ---------------------------------------------------------------------------
// Packing / conversion kernels
// ---------------------------------------------------------------------------
__global__ void k_f32_to_bf16(unsigned short* __restrict__ dst,
                              const float* __restrict__ src, int n) {
    int i = blockIdx.x * 256 + threadIdx.x;
    if (i < n) dst[i] = f32_to_bf16(src[i]);
}

// dst [3072][512] bf16, transposed packing of [W_i | W]  (W_i: [512,2048], W: [512,1024])
__global__ void k_pack_wxt(unsigned short* __restrict__ dst,
                           const float* __restrict__ Wi,
                           const float* __restrict__ Wm) {
    int idx = blockIdx.x * 256 + threadIdx.x;      // 3072*512
    int n = idx >> 9;                               // /512
    int k = idx & 511;
    float v = (n < H2) ? Wi[(size_t)k * H2 + n]
                       : Wm[(size_t)k * HH + (n - H2)];
    dst[idx] = f32_to_bf16(v);
}

// dst [3072][1024] bf16, transposed packing of [W_h | U] (W_h: [1024,2048], U: [1024,1024])
__global__ void k_pack_wht(unsigned short* __restrict__ dst,
                           const float* __restrict__ Wh,
                           const float* __restrict__ Um) {
    int idx = blockIdx.x * 256 + threadIdx.x;      // 3072*1024
    int n = idx >> 10;                              // /1024
    int k = idx & 1023;
    float v = (n < H2) ? Wh[(size_t)k * H2 + n]
                       : Um[(size_t)k * HH + (n - H2)];
    dst[idx] = f32_to_bf16(v);
}

__global__ void k_zero_h(float* __restrict__ hf, unsigned short* __restrict__ hb) {
    int i = blockIdx.x * 256 + threadIdx.x;        // B*H
    hf[i] = 0.0f;
    hb[i] = 0;
}

// ---------------------------------------------------------------------------
// WMMA bf16 GEMM with double-buffered LDS fed by async LDS copies.
//   C[M,N] f32 = A[M,K] bf16 (row-major) x Bt[N,K] bf16 (row-major, = B^T)
// Block tile BM x 128, 256 threads (8 waves).
//   BM=128: waves 2x4, each wave 64x32 (4x2 WMMA 16x16x32)  -> big GEMM
//   BM=64 : waves 1x8, each wave 64x16 (4x1 WMMA 16x16x32)  -> step GEMM
// Tile staging: GLOBAL_LOAD_ASYNC_TO_LDS_B128 (ASYNCcnt), one instruction
// moves 512B (8 rows of 64B); per-wave issue count is PERWAVE per K-tile.
// ---------------------------------------------------------------------------
#define TILE_K     32
#define LDS_STRIDE 40   // 32 data + 8 pad bf16 elems -> 80B rows, 16B aligned

template<int BM>
__global__ __launch_bounds__(256) void gemm_bf16_async(
    const unsigned short* __restrict__ A,
    const unsigned short* __restrict__ Bt,
    float* __restrict__ C,
    int M, int N, int K)
{
    constexpr int WM    = BM / 64;          // waves along M: 2 or 1
    constexpr int WN    = 8 / WM;           // waves along N: 4 or 8
    constexpr int MFRAG = 4;                // 16-row frags per wave (64 rows)
    constexpr int NFRAG = WM;               // 16-col frags per wave (2 or 1)
    constexpr int NSPAN = 16 * NFRAG;       // N cols per wave
    constexpr int AROWS = BM;
    constexpr int BROWS = 128;
    constexpr int AINSTR = AROWS / 8;       // async b128 instrs per A tile
    constexpr int BINSTR = BROWS / 8;       // async b128 instrs per B tile
    constexpr int PERWAVE = (AINSTR + BINSTR) / 8;  // 4 (BM=128) or 3 (BM=64)
    constexpr int ABUF = AROWS * LDS_STRIDE;        // elems per A buffer
    constexpr int BBUF = BROWS * LDS_STRIDE;

    __shared__ __align__(16) unsigned short ldsA[2][ABUF];
    __shared__ __align__(16) unsigned short ldsB[2][BBUF];

    const int tid   = threadIdx.x;
    const int lane  = tid & 31;
    const int wave  = tid >> 5;
    const int waveM = wave / WN;
    const int waveN = wave % WN;
    const int half  = lane >> 4;
    const int lmod  = lane & 15;

    const int blockM = blockIdx.y * BM;
    const int blockN = blockIdx.x * 128;

    // per-lane geometry for async tile staging: 8 rows per instruction,
    // 4 lanes per 64B row, 16B per lane
    const int lrow = lane >> 2;             // 0..7  (row within 8-row slab)
    const int lcol = (lane & 3) * 16;       // byte column 0/16/32/48

    const unsigned ldsAbase = (unsigned)(uintptr_t)&ldsA[0][0];
    const unsigned ldsBbase = (unsigned)(uintptr_t)&ldsB[0][0];

    v8f acc[MFRAG][NFRAG];
    #pragma unroll
    for (int mi = 0; mi < MFRAG; ++mi)
        #pragma unroll
        for (int ni = 0; ni < NFRAG; ++ni)
            acc[mi][ni] = (v8f){};

    const int KT = K / TILE_K;

    // ---- issue helper (async copy one K-tile into LDS buffer `buf`) ----
    auto issue_tile = [&](int kt, int buf) {
        const int k0 = kt * TILE_K;
        // A tile: AROWS x 32
        #pragma unroll
        for (int ii = 0; ii < AINSTR / 8; ++ii) {
            int slab = wave + ii * 8;                    // instruction index
            int row  = slab * 8 + lrow;
            unsigned loff = ldsAbase + (unsigned)(buf * ABUF * 2)
                          + (unsigned)(row * (LDS_STRIDE * 2)) + (unsigned)lcol;
            unsigned long long ga = (unsigned long long)(uintptr_t)A
                + (((unsigned long long)(blockM + row) * (unsigned)K + (unsigned)k0) << 1)
                + (unsigned)lcol;
            asm volatile("global_load_async_to_lds_b128 %0, %1, off"
                         :: "v"(loff), "v"(ga) : "memory");
        }
        // B tile: 128 x 32
        #pragma unroll
        for (int ii = 0; ii < BINSTR / 8; ++ii) {
            int slab = wave + ii * 8;
            int row  = slab * 8 + lrow;
            unsigned loff = ldsBbase + (unsigned)(buf * BBUF * 2)
                          + (unsigned)(row * (LDS_STRIDE * 2)) + (unsigned)lcol;
            unsigned long long gb = (unsigned long long)(uintptr_t)Bt
                + (((unsigned long long)(blockN + row) * (unsigned)K + (unsigned)k0) << 1)
                + (unsigned)lcol;
            asm volatile("global_load_async_to_lds_b128 %0, %1, off"
                         :: "v"(loff), "v"(gb) : "memory");
        }
    };

    issue_tile(0, 0);

    for (int kt = 0; kt < KT; ++kt) {
        const int cur = kt & 1;
        if (kt + 1 < KT) {
            issue_tile(kt + 1, cur ^ 1);
            // current tile's copies (issued earlier, in-order) are complete
            // once only the next tile's PERWAVE copies remain outstanding
            asm volatile("s_wait_asynccnt %0" :: "i"(PERWAVE) : "memory");
        } else {
            asm volatile("s_wait_asynccnt %0" :: "i"(0) : "memory");
        }
        __syncthreads();   // all waves' copies for tile kt visible in LDS

        const unsigned short* bufA = &ldsA[cur][0];
        const unsigned short* bufB = &ldsB[cur][0];

        // A frags: lanes 0-15 hold K 0-7/16-23; lanes 16-31 hold 8-15/24-31
        ABFrag afr[MFRAG];
        #pragma unroll
        for (int mi = 0; mi < MFRAG; ++mi) {
            int row = waveM * 64 + mi * 16 + lmod;
            afr[mi].q[0] = *(const uint4*)&bufA[row * LDS_STRIDE + 8 * half];
            afr[mi].q[1] = *(const uint4*)&bufA[row * LDS_STRIDE + 16 + 8 * half];
        }
        // B frags: lanes 0-15 hold K 0-7/16-23; lanes 16-31 hold 8-15/24-31
        ABFrag bfr[NFRAG];
        #pragma unroll
        for (int ni = 0; ni < NFRAG; ++ni) {
            int row = waveN * NSPAN + ni * 16 + lmod;
            bfr[ni].q[0] = *(const uint4*)&bufB[row * LDS_STRIDE + 8 * half];
            bfr[ni].q[1] = *(const uint4*)&bufB[row * LDS_STRIDE + 16 + 8 * half];
        }

        #pragma unroll
        for (int mi = 0; mi < MFRAG; ++mi)
            #pragma unroll
            for (int ni = 0; ni < NFRAG; ++ni)
                acc[mi][ni] = __builtin_amdgcn_wmma_f32_16x16x32_bf16(
                    false, afr[mi].v, false, bfr[ni].v,
                    (short)0, acc[mi][ni], false, false);

        __syncthreads();   // all waves done reading buf[cur] before reuse
    }

    // C layout: VGPR r -> M = r + 8*half, N = lane&15
    #pragma unroll
    for (int mi = 0; mi < MFRAG; ++mi) {
        #pragma unroll
        for (int ni = 0; ni < NFRAG; ++ni) {
            int ncol = blockN + waveN * NSPAN + ni * 16 + lmod;
            #pragma unroll
            for (int r = 0; r < 8; ++r) {
                int mrow = blockM + waveM * 64 + mi * 16 + r + 8 * half;
                C[(size_t)mrow * N + ncol] = acc[mi][ni][r];
            }
        }
    }
}

// ---------------------------------------------------------------------------
// In-place LN of the x projection: row = [B*T, 3072]
//   cols [0,2048):  LN2 (g2,b2)   cols [2048,3072): LN3 (g3,b3)
// ---------------------------------------------------------------------------
__global__ __launch_bounds__(256) void k_ln_x(
    float* __restrict__ xproj,
    const float* __restrict__ g2, const float* __restrict__ b2,
    const float* __restrict__ g3, const float* __restrict__ b3)
{
    __shared__ float sred[8];
    float* row = xproj + (size_t)blockIdx.x * H3;
    const int tid = threadIdx.x;

    float v1[8], v2[4];
    float s = 0.f, sq = 0.f;
    #pragma unroll
    for (int i = 0; i < 8; ++i) {
        v1[i] = row[tid + i * 256];
        s += v1[i]; sq += v1[i] * v1[i];
    }
    float tot  = block_sum(s, sred);
    float totq = block_sum(sq, sred);
    float m1 = tot * (1.0f / H2);
    float r1 = rsqrtf(totq * (1.0f / H2) - m1 * m1 + 1e-5f);

    s = 0.f; sq = 0.f;
    #pragma unroll
    for (int i = 0; i < 4; ++i) {
        v2[i] = row[H2 + tid + i * 256];
        s += v2[i]; sq += v2[i] * v2[i];
    }
    float t2  = block_sum(s, sred);
    float t2q = block_sum(sq, sred);
    float m2 = t2 * (1.0f / HH);
    float r2 = rsqrtf(t2q * (1.0f / HH) - m2 * m2 + 1e-5f);

    #pragma unroll
    for (int i = 0; i < 8; ++i) {
        int j = tid + i * 256;
        row[j] = (v1[i] - m1) * r1 * g2[j] + b2[j];
    }
    #pragma unroll
    for (int i = 0; i < 4; ++i) {
        int j = tid + i * 256;
        row[H2 + j] = (v2[i] - m2) * r2 * g3[j] + b3[j];
    }
}

// ---------------------------------------------------------------------------
// Per-step fused LN1/LN4 + gates + state update. One block per batch row.
// ---------------------------------------------------------------------------
__global__ __launch_bounds__(256) void k_step(
    const float* __restrict__ G,
    const float* __restrict__ xproj,
    float* __restrict__ hf,
    unsigned short* __restrict__ hb,
    const float* __restrict__ g1, const float* __restrict__ b1,
    const float* __restrict__ g4, const float* __restrict__ b4,
    float* __restrict__ out, int t)
{
    __shared__ float sred[8];
    const int b   = blockIdx.x;
    const int tid = threadIdx.x;
    const float* Gr = G + (size_t)b * H3;
    const float* xp = xproj + ((size_t)b * TT + t) * H3;

    float gz[8], gu[4];
    float s = 0.f, sq = 0.f;
    #pragma unroll
    for (int i = 0; i < 8; ++i) {
        gz[i] = Gr[tid + i * 256];
        s += gz[i]; sq += gz[i] * gz[i];
    }
    float tot  = block_sum(s, sred);
    float totq = block_sum(sq, sred);
    float m1 = tot * (1.0f / H2);
    float r1 = rsqrtf(totq * (1.0f / H2) - m1 * m1 + 1e-5f);

    s = 0.f; sq = 0.f;
    #pragma unroll
    for (int i = 0; i < 4; ++i) {
        gu[i] = Gr[H2 + tid + i * 256];
        s += gu[i]; sq += gu[i] * gu[i];
    }
    float t2  = block_sum(s, sred);
    float t2q = block_sum(sq, sred);
    float m2 = t2 * (1.0f / HH);
    float r2 = rsqrtf(t2q * (1.0f / HH) - m2 * m2 + 1e-5f);

    #pragma unroll
    for (int i = 0; i < 4; ++i) {
        int j = tid + i * 256;                   // output index 0..1023
        float z  = (gz[i]     - m1) * r1 * g1[j]      + b1[j]      + xp[j];
        float rr = (gz[i + 4] - m1) * r1 * g1[j + HH] + b1[j + HH] + xp[j + HH];
        float u  = (gu[i]     - m2) * r2 * g4[j]      + b4[j];
        float hv = hf[(size_t)b * HH + j];
        float hh = tanhf(xp[H2 + j] + sigmoidf_(rr) * u);
        float zs = sigmoidf_(z);
        float hn = (1.0f - zs) * hv + zs * hh;
        hf[(size_t)b * HH + j] = hn;
        hb[(size_t)b * HH + j] = f32_to_bf16(hn);
        out[((size_t)b * TT + t) * HH + j] = hn;             // Hseq [B,T,H]
        if (t == TT - 1)
            out[(size_t)BB * TT * HH + (size_t)b * HH + j] = hn;  // hT
    }
}

// ---------------------------------------------------------------------------
// Launch
// ---------------------------------------------------------------------------
extern "C" void kernel_launch(void* const* d_in, const int* in_sizes, int n_in,
                              void* d_out, int out_size, void* d_ws, size_t ws_size,
                              hipStream_t stream) {
    const float* x    = (const float*)d_in[0];
    const float* W_i  = (const float*)d_in[1];
    const float* W_h  = (const float*)d_in[2];
    const float* Wm   = (const float*)d_in[3];
    const float* Um   = (const float*)d_in[4];
    const float* ln1g = (const float*)d_in[5];
    const float* ln1b = (const float*)d_in[6];
    const float* ln2g = (const float*)d_in[7];
    const float* ln2b = (const float*)d_in[8];
    const float* ln3g = (const float*)d_in[9];
    const float* ln3b = (const float*)d_in[10];
    const float* ln4g = (const float*)d_in[11];
    const float* ln4b = (const float*)d_in[12];
    float* out = (float*)d_out;

    // Workspace layout (bytes)
    char* ws = (char*)d_ws;
    size_t off = 0;
    unsigned short* x_bf  = (unsigned short*)(ws + off); off += (size_t)BT * DD * 2;   // 32 MB
    unsigned short* WxT   = (unsigned short*)(ws + off); off += (size_t)H3 * DD * 2;   // 3 MB
    unsigned short* WhT   = (unsigned short*)(ws + off); off += (size_t)H3 * HH * 2;   // 6 MB
    unsigned short* h_bf  = (unsigned short*)(ws + off); off += (size_t)BB * HH * 2;   // 0.5 MB
    float*          h_f   = (float*)(ws + off);          off += (size_t)BB * HH * 4;   // 1 MB
    float*          G     = (float*)(ws + off);          off += (size_t)BB * H3 * 4;   // 3 MB
    float*          xproj = (float*)(ws + off);          off += (size_t)BT * H3 * 4;   // 384 MB

    // 1) Convert / pack inputs to bf16 (weights transposed to [N,K])
    k_f32_to_bf16<<<(BT * DD) / 256, 256, 0, stream>>>(x_bf, x, BT * DD);
    k_pack_wxt<<<(H3 * DD) / 256, 256, 0, stream>>>(WxT, W_i, Wm);
    k_pack_wht<<<(H3 * HH) / 256, 256, 0, stream>>>(WhT, W_h, Um);
    k_zero_h<<<(BB * HH) / 256, 256, 0, stream>>>(h_f, h_bf);

    // 2) Hoisted input projection: xproj[BT,3072] = x_bf[BT,512] @ [W_i|W]
    {
        dim3 grid(H3 / 128, BT / 128);   // 24 x 256
        gemm_bf16_async<128><<<grid, 256, 0, stream>>>(x_bf, WxT, xproj, BT, H3, DD);
    }
    // 3) LN2 / LN3 in place
    k_ln_x<<<BT, 256, 0, stream>>>(xproj, ln2g, ln2b, ln3g, ln3b);

    // 4) Sequential recurrence: per step, recurrent GEMM + fused gate update
    for (int t = 0; t < TT; ++t) {
        dim3 grid(H3 / 128, BB / 64);    // 24 x 4 = 96 blocks
        gemm_bf16_async<64><<<grid, 256, 0, stream>>>(h_bf, WhT, G, BB, H3, HH);
        k_step<<<BB, 256, 0, stream>>>(G, xproj, h_f, h_bf,
                                       ln1g, ln1b, ln4g, ln4b, out, t);
    }
}
